// AttentionPairBias_11055245820596
// MI455X (gfx1250) — matrix operand order repaired and checked by reading, not compiled, not run on hardware
//
#include <hip/hip_runtime.h>
#include <hip/hip_bf16.h>
#include <stdint.h>

typedef __attribute__((ext_vector_type(16))) _Float16 v16h;
typedef __attribute__((ext_vector_type(8)))  _Float16 v8h;
typedef __attribute__((ext_vector_type(8)))  float    v8f;

constexpr int cB = 2, cN = 512, cC = 768, cH = 16, cD = 48, cCS = 384, cCZ = 128;
constexpr int cSPLIT = 4;                          // flash attention j-splits
constexpr float RSQRT_D = 0.14433756729740643f;    // 1/sqrt(48)

// ---------------- WMMA helpers (CDNA5 16x16x32 f16, f32 accum) ----------------

__device__ __forceinline__ v8f wmma32(v16h a, v16h b, v8f c) {
  return __builtin_amdgcn_wmma_f32_16x16x32_f16(false, a, false, b, (short)0, c, false, false);
}

// A-fragment 16x32 f16: lane L holds row (L&15); lanes>=16 hold K 16..31 (contiguous).
__device__ __forceinline__ v16h load_a_frag(const _Float16* A, int ldk, int row, int kk, int lane) {
  int off = (lane & 16) ? 16 : 0;
  const _Float16* p = A + (size_t)row * ldk + kk + off;
  v8h lo = *(const v8h*)(p);
  v8h hi = *(const v8h*)(p + 8);
  return __builtin_shufflevector(lo, hi, 0,1,2,3,4,5,6,7,8,9,10,11,12,13,14,15);
}

// B-fragment 32x16 f16 for A@W^T (W is [Nout,K] row-major): one contiguous 32B load.
__device__ __forceinline__ v16h load_b_frag(const _Float16* W, int ldk, int col, int kk, int lane) {
  int off = (lane & 16) ? 16 : 0;
  return *(const v16h*)(W + (size_t)col * ldk + kk + off);
}

__device__ __forceinline__ float sigmoidf(float x) { return 1.0f / (1.0f + __expf(-x)); }

// ---------------- weight f32 -> f16 conversion ----------------

__global__ void cvt_kernel(const float* __restrict__ src, _Float16* __restrict__ dst, int n) {
  int i = blockIdx.x * blockDim.x + threadIdx.x;
  if (i < n) dst[i] = (_Float16)src[i];
}

// ---------------- LayerNorms of a_i and s_i (one block per token row) ----------------

__device__ __forceinline__ float block_sum(float v, float* red) {
  int t = threadIdx.x;
  red[t] = v; __syncthreads();
  for (int o = 128; o > 0; o >>= 1) { if (t < o) red[t] += red[t + o]; __syncthreads(); }
  float r = red[0]; __syncthreads();
  return r;
}

__global__ __launch_bounds__(256) void ln_kernel(const float* __restrict__ a_i,
                                                 const float* __restrict__ s_i,
                                                 const float* __restrict__ ada_ln2_w,
                                                 float* __restrict__ aln,
                                                 _Float16* __restrict__ sln_h,
                                                 _Float16* __restrict__ si_h) {
  __shared__ float red[256];
  int row = blockIdx.x, t = threadIdx.x;
  const float* ar = a_i + (size_t)row * cC;
  float s = 0.f;
  for (int i = t; i < cC; i += 256) s += ar[i];
  float mean = block_sum(s, red) * (1.f / cC);
  float vs = 0.f;
  for (int i = t; i < cC; i += 256) { float d = ar[i] - mean; vs += d * d; }
  float rstd = rsqrtf(block_sum(vs, red) * (1.f / cC) + 1e-5f);
  for (int i = t; i < cC; i += 256) aln[(size_t)row * cC + i] = (ar[i] - mean) * rstd;
  const float* sr = s_i + (size_t)row * cCS;
  s = 0.f;
  for (int i = t; i < cCS; i += 256) s += sr[i];
  float ms = block_sum(s, red) * (1.f / cCS);
  vs = 0.f;
  for (int i = t; i < cCS; i += 256) { float d = sr[i] - ms; vs += d * d; }
  float rs = rsqrtf(block_sum(vs, red) * (1.f / cCS) + 1e-5f);
  for (int i = t; i < cCS; i += 256) {
    sln_h[(size_t)row * cCS + i] = (_Float16)((sr[i] - ms) * rs * ada_ln2_w[i]);
    si_h [(size_t)row * cCS + i] = (_Float16)sr[i];
  }
}

// ---------------- AdaLN conditioning ----------------

__global__ __launch_bounds__(256) void cond_kernel(const _Float16* __restrict__ sln,
                                                   const _Float16* __restrict__ Wlh,
                                                   const _Float16* __restrict__ Wnbh,
                                                   const float* __restrict__ bl,
                                                   const float* __restrict__ aln,
                                                   _Float16* __restrict__ acond) {
  const int NT = cC / 16, MT = (cB * cN) / 16;
  int wave = threadIdx.x >> 5, lane = threadIdx.x & 31;
  int tile = blockIdx.x * 8 + wave;
  if (tile >= MT * NT) return;
  int tm = tile / NT, tn = tile % NT;
  int row = tm * 16 + (lane & 15);
  int col = tn * 16 + (lane & 15);
  v8f acc1 = {}, acc2 = {};
  for (int kk = 0; kk < cCS; kk += 32) {
    v16h a  = load_a_frag(sln, cCS, row, kk, lane);
    v16h b1 = load_b_frag(Wlh,  cCS, col, kk, lane);
    v16h b2 = load_b_frag(Wnbh, cCS, col, kk, lane);
    acc1 = wmma32(a, b1, acc1);
    acc2 = wmma32(a, b2, acc2);
  }
  float blc = bl[col];
  int hoff = (lane & 16) ? 8 : 0;
#pragma unroll
  for (int r = 0; r < 8; r++) {
    int rowM = tm * 16 + r + hoff;
    float g = sigmoidf(acc1[r] + blc);
    float v = g * aln[(size_t)rowM * cC + col] + acc2[r];
    acond[(size_t)rowM * cC + col] = (_Float16)v;
  }
}

// ---------------- output gate: gate_s = sigmoid(s_i @ Wgs^T + bgs) ----------------

__global__ __launch_bounds__(256) void gate_kernel(const _Float16* __restrict__ si_h,
                                                   const _Float16* __restrict__ Wgsh,
                                                   const float* __restrict__ bgs,
                                                   float* __restrict__ gate_s) {
  const int NT = cC / 16, MT = (cB * cN) / 16;
  int wave = threadIdx.x >> 5, lane = threadIdx.x & 31;
  int tile = blockIdx.x * 8 + wave;
  if (tile >= MT * NT) return;
  int tm = tile / NT, tn = tile % NT;
  int row = tm * 16 + (lane & 15);
  int col = tn * 16 + (lane & 15);
  v8f acc = {};
  for (int kk = 0; kk < cCS; kk += 32) {
    v16h a = load_a_frag(si_h, cCS, row, kk, lane);
    v16h b = load_b_frag(Wgsh, cCS, col, kk, lane);
    acc = wmma32(a, b, acc);
  }
  float bc = bgs[col];
  int hoff = (lane & 16) ? 8 : 0;
#pragma unroll
  for (int r = 0; r < 8; r++) {
    int rowM = tm * 16 + r + hoff;
    gate_s[(size_t)rowM * cC + col] = sigmoidf(acc[r] + bc);
  }
}

// ---------------- pair bias: bias[b,h,i,j] = LN(z[b,i,j,:]) . Wz[h,:] + beta[b,i,j] ----------------
// One wave per (b,i, 16-wide j tile). The 16x128 f32 z tile is one contiguous 8KB block;
// stage it through LDS with CDNA5 async copies (global_load_async_to_lds_b128, ASYNCcnt),
// then LN in registers with the half-lane stat exchange matching the A-fragment layout.

__global__ __launch_bounds__(128) void bias_kernel(const float* __restrict__ z,
                                                   const float* __restrict__ beta,
                                                   const _Float16* __restrict__ Wzh,
                                                   const float* __restrict__ lnz_w,
                                                   const float* __restrict__ lnz_b,
                                                   _Float16* __restrict__ biasb) {
  __shared__ __align__(32) float ztile[4][16 * cCZ];   // 4 waves x 8KB
  const int NT = cN / 16;
  int wave = threadIdx.x >> 5, lane = threadIdx.x & 31;
  int tile = blockIdx.x * 4 + wave;
  if (tile >= cB * cN * NT) return;
  int jt = tile % NT;
  int bi = tile / NT;                 // (b * N + i)
  int j0 = jt * 16;
  int laneM = lane & 15;
  int off8 = (lane & 16) ? 8 : 0;

  // --- async stage: 2KB contiguous (rows j0..j0+3 of z[bi]) -> LDS slice ---
  float* zbuf = ztile[wave];
  const char* src = (const char*)(z + ((size_t)bi * cN + j0) * cCZ);
  unsigned ldsbase = (unsigned)(uintptr_t)zbuf;
#pragma unroll
  for (int i = 0; i < 4; i++) {
    unsigned boff = (unsigned)((i * 32 + lane) * 16);
    unsigned ldsa = ldsbase + boff;
    uint64_t ga = (uint64_t)(uintptr_t)src + boff;
    asm volatile("global_load_async_to_lds_b128 %0, %1, off"
                 :: "v"(ldsa), "v"(ga) : "memory");
  }
  asm volatile("s_wait_asynccnt 0x0" ::: "memory");

  const float* zr = zbuf + laneM * cCZ;   // this lane's row (shared with lane^16)
  float x[64];
#pragma unroll
  for (int ks = 0; ks < 4; ks++) {
    int base = ks * 32 + 2 * off8;
    const float4* p0 = (const float4*)(zr + base);
    const float4* p1 = (const float4*)(zr + base + 8);
    float4 f0 = p0[0], f1 = p0[1], f2 = p1[0], f3 = p1[1];
    x[ks*16+ 0]=f0.x; x[ks*16+ 1]=f0.y; x[ks*16+ 2]=f0.z; x[ks*16+ 3]=f0.w;
    x[ks*16+ 4]=f1.x; x[ks*16+ 5]=f1.y; x[ks*16+ 6]=f1.z; x[ks*16+ 7]=f1.w;
    x[ks*16+ 8]=f2.x; x[ks*16+ 9]=f2.y; x[ks*16+10]=f2.z; x[ks*16+11]=f2.w;
    x[ks*16+12]=f3.x; x[ks*16+13]=f3.y; x[ks*16+14]=f3.z; x[ks*16+15]=f3.w;
  }
  float s = 0.f, s2 = 0.f;
#pragma unroll
  for (int i = 0; i < 64; i++) { s += x[i]; s2 += x[i] * x[i]; }
  s  += __shfl_xor(s, 16, 32);
  s2 += __shfl_xor(s2, 16, 32);
  float mean = s * (1.f / cCZ);
  float var  = s2 * (1.f / cCZ) - mean * mean;
  float rstd = rsqrtf(var + 1e-5f);

  v8f acc = {};
#pragma unroll
  for (int ks = 0; ks < 4; ks++) {
    int kk = ks * 32;
    v16h af;
#pragma unroll
    for (int e = 0; e < 16; e++) {
      int kidx = kk + 2 * off8 + e;
      float zn = (x[ks * 16 + e] - mean) * rstd * lnz_w[kidx] + lnz_b[kidx];
      af[e] = (_Float16)zn;
    }
    v16h bf = load_b_frag(Wzh, cCZ, laneM, kk, lane);   // head = laneM
    acc = wmma32(af, bf, acc);
  }
  int b = bi / cN, i = bi % cN;
  int h = laneM;
  const float* brow = beta + (size_t)bi * cN;
  _Float16* bout = biasb + (((size_t)(b * cH + h) * cN) + i) * cN;
#pragma unroll
  for (int r = 0; r < 8; r++) {
    int j = j0 + r + off8;
    bout[j] = (_Float16)(acc[r] + brow[j]);
  }
}

// ---------------- fused Q/K/V/G projections ----------------

__global__ __launch_bounds__(256) void qkvg_kernel(const _Float16* __restrict__ acond,
                                                   const _Float16* __restrict__ Wqh,
                                                   const _Float16* __restrict__ Wkh,
                                                   const _Float16* __restrict__ Wvh,
                                                   const _Float16* __restrict__ Wgh,
                                                   const float* __restrict__ bq,
                                                   _Float16* __restrict__ q_h,
                                                   _Float16* __restrict__ k_h,
                                                   _Float16* __restrict__ vT,
                                                   float* __restrict__ gsig) {
  const int NT = cC / 16, MT = (cB * cN) / 16;
  int wave = threadIdx.x >> 5, lane = threadIdx.x & 31;
  int tile = blockIdx.x * 8 + wave;
  if (tile >= MT * NT) return;
  int which = blockIdx.y;
  const _Float16* W = (which == 0) ? Wqh : (which == 1) ? Wkh : (which == 2) ? Wvh : Wgh;
  int tm = tile / NT, tn = tile % NT;
  int row = tm * 16 + (lane & 15);
  int col = tn * 16 + (lane & 15);
  v8f acc = {};
  for (int kk = 0; kk < cC; kk += 32) {
    v16h a = load_a_frag(acond, cC, row, kk, lane);
    v16h b = load_b_frag(W,     cC, col, kk, lane);
    acc = wmma32(a, b, acc);
  }
  int h = col / cD, d = col % cD;
  int hoff = (lane & 16) ? 8 : 0;
#pragma unroll
  for (int r = 0; r < 8; r++) {
    int rowM = tm * 16 + r + hoff;
    int b = rowM >> 9, t = rowM & (cN - 1);
    size_t qk = ((size_t)(b * cH + h) * cN + t) * 64 + d;
    if (which == 0) {
      q_h[qk] = (_Float16)((acc[r] + bq[col]) * RSQRT_D);
      if (d >= 32) q_h[qk + 16] = (_Float16)0.f;        // zero the D=48..63 pad
    } else if (which == 1) {
      k_h[qk] = (_Float16)acc[r];
      if (d >= 32) k_h[qk + 16] = (_Float16)0.f;
    } else if (which == 2) {
      vT[((size_t)(b * cH + h) * cD + d) * cN + t] = (_Float16)acc[r];
    } else {
      gsig[(size_t)rowM * cC + col] = sigmoidf(acc[r]);
    }
  }
}

// ---------------- split-K flash attention: one wave per (split, b, h, 16-row q tile) ----------------
// Each wave covers 128 keys; writes unnormalized partial O plus per-row (m, l).

__global__ __launch_bounds__(256) void attn_kernel(const _Float16* __restrict__ q_h,
                                                   const _Float16* __restrict__ k_h,
                                                   const _Float16* __restrict__ vT,
                                                   const _Float16* __restrict__ biasb,
                                                   float* __restrict__ opart,
                                                   float* __restrict__ mpart,
                                                   float* __restrict__ lpart) {
  __shared__ __align__(32) _Float16 plds[8][16 * 32];
  const int QT = cN / 16, BH = cB * cH;
  int wave = threadIdx.x >> 5, lane = threadIdx.x & 31;
  int tile = blockIdx.x * 8 + wave;
  if (tile >= cSPLIT * BH * QT) return;
  int qt = tile % QT;
  int rest = tile / QT;
  int bh = rest % BH;
  int split = rest / BH;
  int q0 = qt * 16;
  int jbeg = split * (cN / cSPLIT);
  int laneM = lane & 15;
  int hoff = (lane & 16) ? 8 : 0;
  int off16 = (lane & 16) ? 16 : 0;

  const _Float16* qbase = q_h + (size_t)bh * cN * 64;
  const _Float16* kbase = k_h + (size_t)bh * cN * 64;
  const _Float16* vbase = vT + (size_t)bh * cD * cN;
  const _Float16* brow = biasb + ((size_t)bh * cN + q0) * cN;
  _Float16* pbuf = plds[wave];

  v16h qa[2];
#pragma unroll
  for (int kq = 0; kq < 2; kq++)
    qa[kq] = load_a_frag(qbase, 64, q0 + laneM, kq * 32, lane);

  v8f accO0 = {}, accO1 = {}, accO2 = {};
  float m8[8], l8[8];
#pragma unroll
  for (int r = 0; r < 8; r++) { m8[r] = -1e30f; l8[r] = 0.f; }

  for (int j0 = jbeg; j0 < jbeg + cN / cSPLIT; j0 += 32) {
    v8f s0 = {}, s1 = {};
#pragma unroll
    for (int kq = 0; kq < 2; kq++) {
      int kk = kq * 32;
      v16h kb0 = *(const v16h*)(kbase + (size_t)(j0 + laneM) * 64 + kk + off16);
      v16h kb1 = *(const v16h*)(kbase + (size_t)(j0 + 16 + laneM) * 64 + kk + off16);
      s0 = wmma32(qa[kq], kb0, s0);
      s1 = wmma32(qa[kq], kb1, s1);
    }
    float scale_r[8];
#pragma unroll
    for (int r = 0; r < 8; r++) {
      int rowM = r + hoff;
      const _Float16* bp = brow + (size_t)rowM * cN + j0 + laneM;
      float v0 = s0[r] + (float)bp[0];
      float v1 = s1[r] + (float)bp[16];
      float t = fmaxf(v0, v1);
      t = fmaxf(t, __shfl_xor(t, 1, 32));
      t = fmaxf(t, __shfl_xor(t, 2, 32));
      t = fmaxf(t, __shfl_xor(t, 4, 32));
      t = fmaxf(t, __shfl_xor(t, 8, 32));
      float newm = fmaxf(m8[r], t);
      float p0 = __expf(v0 - newm), p1 = __expf(v1 - newm);
      float ss = p0 + p1;
      ss += __shfl_xor(ss, 1, 32);
      ss += __shfl_xor(ss, 2, 32);
      ss += __shfl_xor(ss, 4, 32);
      ss += __shfl_xor(ss, 8, 32);
      float corr = __expf(m8[r] - newm);
      l8[r] = l8[r] * corr + ss;
      m8[r] = newm;
      scale_r[r] = corr;
      pbuf[rowM * 32 + laneM]      = (_Float16)p0;
      pbuf[rowM * 32 + 16 + laneM] = (_Float16)p1;
    }
    asm volatile("s_wait_dscnt 0x0" ::: "memory");   // cross-lane LDS transpose visibility (per-wave)
#pragma unroll
    for (int r = 0; r < 8; r++) {
      accO0[r] *= scale_r[r];
      accO1[r] *= scale_r[r];
      accO2[r] *= scale_r[r];
    }
    v16h pa = load_a_frag(pbuf, 32, laneM, 0, lane);
    v16h vb0 = *(const v16h*)(vbase + (size_t)(0 * 16 + laneM) * cN + j0 + off16);
    v16h vb1 = *(const v16h*)(vbase + (size_t)(1 * 16 + laneM) * cN + j0 + off16);
    v16h vb2 = *(const v16h*)(vbase + (size_t)(2 * 16 + laneM) * cN + j0 + off16);
    accO0 = wmma32(pa, vb0, accO0);
    accO1 = wmma32(pa, vb1, accO1);
    accO2 = wmma32(pa, vb2, accO2);
  }
  // partial outputs (unnormalized) + per-row stats
  size_t sb = (size_t)(split * BH + bh);
#pragma unroll
  for (int r = 0; r < 8; r++) {
    int rowM = r + hoff;
    size_t orow = (sb * cN + q0 + rowM) * cD;
    opart[orow + 0 * 16 + laneM] = accO0[r];
    opart[orow + 1 * 16 + laneM] = accO1[r];
    opart[orow + 2 * 16 + laneM] = accO2[r];
  }
  if (laneM == 0) {
#pragma unroll
    for (int r = 0; r < 8; r++) {
      int rowM = r + hoff;
      mpart[sb * cN + q0 + rowM] = m8[r];
      lpart[sb * cN + q0 + rowM] = l8[r];
    }
  }
}

// ---------------- combine splits + apply sigmoid(g) gate -> o_h (f16) ----------------

__global__ __launch_bounds__(256) void combine_kernel(const float* __restrict__ opart,
                                                      const float* __restrict__ mpart,
                                                      const float* __restrict__ lpart,
                                                      const float* __restrict__ gsig,
                                                      _Float16* __restrict__ o_h) {
  const int BH = cB * cH;
  int idx = blockIdx.x * 256 + threadIdx.x;
  const int total = BH * cN * cD;
  if (idx >= total) return;
  int d = idx % cD;
  int rest = idx / cD;
  int q = rest % cN;
  int bh = rest / cN;
  float ms[cSPLIT];
  float M = -1e30f;
#pragma unroll
  for (int s = 0; s < cSPLIT; s++) {
    ms[s] = mpart[(size_t)(s * BH + bh) * cN + q];
    M = fmaxf(M, ms[s]);
  }
  float L = 0.f, O = 0.f;
#pragma unroll
  for (int s = 0; s < cSPLIT; s++) {
    float w = __expf(ms[s] - M);
    L += w * lpart[(size_t)(s * BH + bh) * cN + q];
    O += w * opart[((size_t)(s * BH + bh) * cN + q) * cD + d];
  }
  int b = bh / cH, h = bh % cH;
  int tok = b * cN + q;
  int col = h * cD + d;
  float g = gsig[(size_t)tok * cC + col];
  o_h[(size_t)tok * cC + col] = (_Float16)(O / L * g);
}

// ---------------- output projection: out = (g*o) @ Wo^T * gate_s ----------------

__global__ __launch_bounds__(256) void out_kernel(const _Float16* __restrict__ o_h,
                                                  const _Float16* __restrict__ Woh,
                                                  const float* __restrict__ gate_s,
                                                  float* __restrict__ out) {
  const int NT = cC / 16, MT = (cB * cN) / 16;
  int wave = threadIdx.x >> 5, lane = threadIdx.x & 31;
  int tile = blockIdx.x * 8 + wave;
  if (tile >= MT * NT) return;
  int tm = tile / NT, tn = tile % NT;
  int row = tm * 16 + (lane & 15);
  int col = tn * 16 + (lane & 15);
  v8f acc = {};
  for (int kk = 0; kk < cC; kk += 32) {
    v16h a = load_a_frag(o_h, cC, row, kk, lane);
    v16h b = load_b_frag(Woh, cC, col, kk, lane);
    acc = wmma32(a, b, acc);
  }
  int hoff = (lane & 16) ? 8 : 0;
#pragma unroll
  for (int r = 0; r < 8; r++) {
    int rowM = tm * 16 + r + hoff;
    out[(size_t)rowM * cC + col] = acc[r] * gate_s[(size_t)rowM * cC + col];
  }
}

// ---------------- host ----------------

extern "C" void kernel_launch(void* const* d_in, const int* in_sizes, int n_in,
                              void* d_out, int out_size, void* d_ws, size_t ws_size,
                              hipStream_t stream) {
  const float* a_i     = (const float*)d_in[0];
  const float* s_i     = (const float*)d_in[1];
  const float* z_ij    = (const float*)d_in[2];
  const float* beta_ij = (const float*)d_in[3];
  const float* Wq      = (const float*)d_in[4];
  const float* bq      = (const float*)d_in[5];
  const float* Wk      = (const float*)d_in[6];
  const float* Wv      = (const float*)d_in[7];
  const float* Wg      = (const float*)d_in[8];
  const float* Wo      = (const float*)d_in[9];
  const float* ada_ln2_w = (const float*)d_in[10];
  const float* ada_Wl  = (const float*)d_in[11];
  const float* ada_bl  = (const float*)d_in[12];
  const float* ada_Wnb = (const float*)d_in[13];
  const float* lnz_w   = (const float*)d_in[14];
  const float* lnz_b   = (const float*)d_in[15];
  const float* Wz      = (const float*)d_in[16];
  const float* Wgs     = (const float*)d_in[17];
  const float* bgs     = (const float*)d_in[18];
  float* out = (float*)d_out;

  char* w = (char*)d_ws;
  auto alloc = [&](size_t bytes) -> char* {
    char* p = w;
    w += (bytes + 255) & ~(size_t)255;
    return p;
  };
  const size_t nTok = (size_t)cB * cN;
  const int BH = cB * cH;
  _Float16* Wqh  = (_Float16*)alloc((size_t)cC * cC * 2);
  _Float16* Wkh  = (_Float16*)alloc((size_t)cC * cC * 2);
  _Float16* Wvh  = (_Float16*)alloc((size_t)cC * cC * 2);
  _Float16* Wgh  = (_Float16*)alloc((size_t)cC * cC * 2);
  _Float16* Woh  = (_Float16*)alloc((size_t)cC * cC * 2);
  _Float16* Wlh  = (_Float16*)alloc((size_t)cC * cCS * 2);
  _Float16* Wnbh = (_Float16*)alloc((size_t)cC * cCS * 2);
  _Float16* Wgsh = (_Float16*)alloc((size_t)cC * cCS * 2);
  _Float16* Wzh  = (_Float16*)alloc((size_t)cH * cCZ * 2);
  float*    aln    = (float*)   alloc(nTok * cC * 4);
  _Float16* sln_h  = (_Float16*)alloc(nTok * cCS * 2);
  _Float16* si_h   = (_Float16*)alloc(nTok * cCS * 2);
  _Float16* acond  = (_Float16*)alloc(nTok * cC * 2);
  float*    gate_s = (float*)   alloc(nTok * cC * 4);
  float*    gsig   = (float*)   alloc(nTok * cC * 4);
  _Float16* q_h    = (_Float16*)alloc((size_t)cB * cH * cN * 64 * 2);
  _Float16* k_h    = (_Float16*)alloc((size_t)cB * cH * cN * 64 * 2);
  _Float16* vT     = (_Float16*)alloc((size_t)cB * cH * cD * cN * 2);
  _Float16* biasb  = (_Float16*)alloc((size_t)cB * cH * cN * cN * 2);
  float*    opart  = (float*)   alloc((size_t)cSPLIT * BH * cN * cD * 4);
  float*    mpart  = (float*)   alloc((size_t)cSPLIT * BH * cN * 4);
  float*    lpart  = (float*)   alloc((size_t)cSPLIT * BH * cN * 4);
  _Float16* o_h    = (_Float16*)alloc(nTok * cC * 2);

  auto cvt = [&](const float* src, _Float16* dst, int n) {
    cvt_kernel<<<(n + 255) / 256, 256, 0, stream>>>(src, dst, n);
  };
  cvt(Wq, Wqh, cC * cC);
  cvt(Wk, Wkh, cC * cC);
  cvt(Wv, Wvh, cC * cC);
  cvt(Wg, Wgh, cC * cC);
  cvt(Wo, Woh, cC * cC);
  cvt(ada_Wl, Wlh, cC * cCS);
  cvt(ada_Wnb, Wnbh, cC * cCS);
  cvt(Wgs, Wgsh, cC * cCS);
  cvt(Wz, Wzh, cH * cCZ);

  ln_kernel<<<cB * cN, 256, 0, stream>>>(a_i, s_i, ada_ln2_w, aln, sln_h, si_h);

  const int gemmTiles = (int)(nTok / 16) * (cC / 16);          // 3072
  cond_kernel<<<(gemmTiles + 7) / 8, 256, 0, stream>>>(sln_h, Wlh, Wnbh, ada_bl, aln, acond);
  gate_kernel<<<(gemmTiles + 7) / 8, 256, 0, stream>>>(si_h, Wgsh, bgs, gate_s);

  const int biasTiles = cB * cN * (cN / 16);                   // 32768
  bias_kernel<<<(biasTiles + 3) / 4, 128, 0, stream>>>(z_ij, beta_ij, Wzh, lnz_w, lnz_b, biasb);

  qkvg_kernel<<<dim3((gemmTiles + 7) / 8, 4), 256, 0, stream>>>(acond, Wqh, Wkh, Wvh, Wgh, bq,
                                                                q_h, k_h, vT, gsig);

  const int attnTiles = cSPLIT * BH * (cN / 16);               // 4096
  attn_kernel<<<(attnTiles + 7) / 8, 256, 0, stream>>>(q_h, k_h, vT, biasb, opart, mpart, lpart);

  const int combTotal = BH * cN * cD;                          // 786432
  combine_kernel<<<(combTotal + 255) / 256, 256, 0, stream>>>(opart, mpart, lpart, gsig, o_h);

  out_kernel<<<(gemmTiles + 7) / 8, 256, 0, stream>>>(o_h, Woh, gate_s, out);

  (void)in_sizes; (void)n_in; (void)out_size; (void)ws_size;
}
